// Latex_arch_4518305595488
// MI455X (gfx1250) — matrix operand order repaired and checked by reading, not compiled, not run on hardware
//
#include <hip/hip_runtime.h>

typedef unsigned short u16;
typedef unsigned int   u32;
typedef __attribute__((ext_vector_type(16))) __bf16 v16bf;
typedef __attribute__((ext_vector_type(8)))  float  v8f;
typedef __attribute__((ext_vector_type(4)))  u32    u32x4;

union Frag { v16bf v; u32x4 q[2]; };

__device__ __forceinline__ float bf2f(u16 b) {
  u32 u = ((u32)b) << 16; float f; __builtin_memcpy(&f, &u, 4); return f;
}
__device__ __forceinline__ u16 f2bf(float f) {
  u32 u; __builtin_memcpy(&u, &f, 4);
  u32 r = u + 0x7FFFu + ((u >> 16) & 1u);
  return (u16)(r >> 16);
}
__device__ __forceinline__ float sigf(float x) { return 1.f / (1.f + __expf(-x)); }

// A-matrix 16x32 bf16 fragment: lane<16 holds K[0..7] (v0-3) and K[16..23] (v4-7),
// lane>=16 holds K[8..15] and K[24..31]. base = &A[row, k0] (per-lane row).
__device__ __forceinline__ void load_a_frag(Frag& f, const u16* base, int lane) {
  const u16* p = base + ((lane >> 4) * 8);
  f.q[0] = *(const u32x4*)p;
  f.q[1] = *(const u32x4*)(p + 16);
}
// B-matrix 32x16 bf16 fragment: lane<16 holds K[0..7] and K[16..23], lane>=16 K[8..15] and K[24..31].
// base = &Bw[col, k0] (per-lane col, [N,K] row-major).
__device__ __forceinline__ void load_b_frag(Frag& f, const u16* base, int lane) {
  const u16* p = base + ((lane >> 4) * 8);
  f.q[0] = *(const u32x4*)p;
  f.q[1] = *(const u32x4*)(p + 16);
}

#define WMMA_BF16(acc, a, b) \
  acc = __builtin_amdgcn_wmma_f32_16x16x32_bf16(false, (a).v, false, (b).v, (short)0, acc, false, false)

// ---------------------------------------------------------------- prep kernels
__global__ void k_f32_to_bf16(const float* __restrict__ s, u16* __restrict__ d, int n) {
  int i = blockIdx.x * blockDim.x + threadIdx.x;
  if (i < n) d[i] = f2bf(s[i]);
}
// OIHW fp32 -> [O, 25*Cin] bf16 with k = (kh*5+kw)*Cin + ci  (O-HWI)
__global__ void k_convw(const float* __restrict__ s, u16* __restrict__ d, int Cin, int total) {
  int i = blockIdx.x * blockDim.x + threadIdx.x;
  if (i >= total) return;
  int K = 25 * Cin;
  int o = i / K, r = i % K;
  int p = r / Cin, ci = r % Cin;
  d[i] = f2bf(s[(o * Cin + ci) * 25 + p]);
}
__global__ void k_woutpad(const float* __restrict__ s, u16* __restrict__ d) {
  int i = blockIdx.x * blockDim.x + threadIdx.x;
  if (i >= 608 * 512) return;
  int row = i >> 9, col = i & 511;
  d[i] = (row < 600) ? f2bf(s[row * 512 + col]) : (u16)0;
}
__global__ void k_biascomb(const float* __restrict__ a, const float* __restrict__ b,
                           float* __restrict__ d) {
  int i = blockIdx.x * blockDim.x + threadIdx.x;
  if (i < 2048) d[i] = a[i] + b[i];
}
__global__ void k_boutpad(const float* __restrict__ s, float* __restrict__ d) {
  int i = blockIdx.x * blockDim.x + threadIdx.x;
  if (i < 608) d[i] = (i < 600) ? s[i] : -1e30f;
}
__global__ void k_zero_step0(float* __restrict__ out) {
  int i = blockIdx.x * blockDim.x + threadIdx.x;
  if (i >= 64 * 600) return;
  int b = i / 600, v = i % 600;
  out[(long)b * 390000 + v] = 0.f;
}

// --------------------------------------- conv1 (Cin=3) direct, fused 2x2 maxpool
// x: NCHW fp32 [64,3,224,224]; out: NHWC bf16 [64,110,110,32]
__global__ void k_conv1_pool(const float* __restrict__ x, const float* __restrict__ w,
                             const float* __restrict__ bias, u16* __restrict__ out) {
  __shared__ float wsh[2400]; // [32][75] == OIHW flat
  for (int i = threadIdx.x; i < 2400; i += blockDim.x) wsh[i] = w[i];
  __syncthreads();
  int t = blockIdx.x * blockDim.x + threadIdx.x;
  const int total = 64 * 110 * 110;
  if (t >= total) return;
  int b = t / (110 * 110), r = t % (110 * 110);
  int ph = r / 110, pw = r % 110;
  float pooled[32];
#pragma unroll
  for (int co = 0; co < 32; ++co) pooled[co] = 0.f;
  for (int pos = 0; pos < 4; ++pos) {
    int oh = 2 * ph + (pos >> 1), ow = 2 * pw + (pos & 1);
    float patch[75];
#pragma unroll
    for (int ci = 0; ci < 3; ++ci)
#pragma unroll
      for (int kh = 0; kh < 5; ++kh)
#pragma unroll
        for (int kw = 0; kw < 5; ++kw)
          patch[ci * 25 + kh * 5 + kw] =
              x[(((long)b * 3 + ci) * 224 + oh + kh) * 224 + ow + kw];
    for (int co = 0; co < 32; ++co) {
      float a = bias[co];
#pragma unroll
      for (int k = 0; k < 75; ++k) a += patch[k] * wsh[co * 75 + k];
      a = a > 0.f ? a : 0.f;
      if (a > pooled[co]) pooled[co] = a;
    }
  }
  u16* op = out + (((long)b * 110 + ph) * 110 + pw) * 32;
#pragma unroll
  for (int co = 0; co < 32; ++co) op[co] = f2bf(pooled[co]);
}

// ------------------------------- implicit-GEMM WMMA conv (layers 2..5), NHWC bf16
// Compile-time shapes: divisions fold to mul-shift, K-loop trip count constant.
// in [B,H,W,Cin], wt [Cout, 25*Cin] (O-HWI), out [B,OH,OW,Cout]; bias+relu fused.
// grid = (M/64, Cout/64), block 256 (8 waves: 4 m-tiles x 2 n-groups).
// Each wave computes a 16x32 output tile (2 accumulators, A-fragment reused 2x).
template <int H, int W, int Cin, int Cout>
__global__ void k_conv_wmma(const u16* __restrict__ in, const u16* __restrict__ wt,
                            const float* __restrict__ bias, u16* __restrict__ out) {
  constexpr int OH = H - 4, OW = W - 4;
  constexpr int K  = 25 * Cin;
  const int lane = threadIdx.x & 31;
  const int wave = threadIdx.x >> 5;
  const int m0 = blockIdx.x * 64 + (wave >> 1) * 16;
  const int n0 = blockIdx.y * 64 + (wave & 1) * 32;
  const int mrow = m0 + (lane & 15);
  const int b  = mrow / (OH * OW);
  const int rr = mrow % (OH * OW);
  const int oh = rr / OW, ow = rr % OW;
  const long inbase = (((long)b * H + oh) * W + ow) * Cin;
  const int ncol0 = n0 + (lane & 15);
  const u16* wrow0 = wt + (long)ncol0 * K;
  const u16* wrow1 = wrow0 + (long)16 * K;
  v8f acc0 = {0, 0, 0, 0, 0, 0, 0, 0};
  v8f acc1 = {0, 0, 0, 0, 0, 0, 0, 0};
  for (int k0 = 0; k0 < K; k0 += 32) {
    const int p  = k0 / Cin;          // Cin multiple of 32 -> chunk stays in one patch
    const int ci = k0 % Cin;
    const int kh = p / 5, kw = p % 5;
    Frag a, b0, b1;
    load_a_frag(a, in + inbase + ((long)kh * W + kw) * Cin + ci, lane);
    load_b_frag(b0, wrow0 + k0, lane);
    load_b_frag(b1, wrow1 + k0, lane);
    WMMA_BF16(acc0, a, b0);
    WMMA_BF16(acc1, a, b1);
  }
  const float bv0 = bias[ncol0];
  const float bv1 = bias[ncol0 + 16];
  const int rbase = (lane >> 4) * 8;
#pragma unroll
  for (int i = 0; i < 8; ++i) {
    int m = m0 + rbase + i;
    int b2 = m / (OH * OW), r2 = m % (OH * OW);
    u16* orow = out + (((long)b2 * OH + (r2 / OW)) * OW + (r2 % OW)) * Cout;
    float v0 = acc0[i] + bv0; v0 = v0 > 0.f ? v0 : 0.f;
    float v1 = acc1[i] + bv1; v1 = v1 > 0.f ? v1 : 0.f;
    orow[ncol0]      = f2bf(v0);
    orow[ncol0 + 16] = f2bf(v1);
  }
}

// 2x2 maxpool, NHWC bf16
__global__ void k_pool(const u16* __restrict__ in, u16* __restrict__ out,
                       int H, int W, int C, int PH, int PW, int total) {
  int i = blockIdx.x * blockDim.x + threadIdx.x;
  if (i >= total) return;
  int c = i % C, r = i / C;
  int pw = r % PW; r /= PW;
  int ph = r % PH; int b = r / PH;
  const u16* p = in + (((long)b * H + 2 * ph) * W + 2 * pw) * C + c;
  float v0 = bf2f(p[0]), v1 = bf2f(p[C]);
  float v2 = bf2f(p[(long)W * C]), v3 = bf2f(p[(long)W * C + C]);
  out[i] = f2bf(fmaxf(fmaxf(v0, v1), fmaxf(v2, v3)));
}

// ctx = mean over 3x3 of relu(pool5) ; pool5 NHWC bf16 [64,3,3,512]
__global__ void k_ctx(const u16* __restrict__ p5, float* __restrict__ ctx) {
  int i = blockIdx.x * blockDim.x + threadIdx.x;
  if (i >= 64 * 512) return;
  int b = i >> 9, ch = i & 511;
  float s = 0.f;
#pragma unroll
  for (int r = 0; r < 3; ++r)
#pragma unroll
    for (int cc = 0; cc < 3; ++cc) {
      float v = bf2f(p5[(((long)b * 3 + r) * 3 + cc) * 512 + ch]);
      s += (v > 0.f ? v : 0.f);
    }
  ctx[i] = s * (1.f / 9.f);
}

// init LSTM carry: xt = [ctx | emb[START=1]], h = ctx(bf16), c = ctx(f32)
__global__ void k_init(const float* __restrict__ ctx, const u16* __restrict__ embbf,
                       u16* __restrict__ xt, u16* __restrict__ h, float* __restrict__ c) {
  int i = blockIdx.x * blockDim.x + threadIdx.x;
  if (i >= 64 * 512) return;
  int b = i >> 9, n = i & 511;
  float cv = ctx[i];
  u16 cb = f2bf(cv);
  xt[b * 1024 + n]       = cb;
  xt[b * 1024 + 512 + n] = embbf[512 + n];  // token 1
  h[i] = cb;
  c[i] = cv;
}

// ------------------------------------------------- persistent LSTM decode kernel
// 16 WGs x 256 threads = 128 waves. Grid-wide epoch barrier between phases.
__device__ __forceinline__ void grid_sync(u32* cnt, u32* epoch, int nwg) {
  __syncthreads();
  if (threadIdx.x == 0) {
    __threadfence();
    u32 e = ++(*epoch);
    atomicAdd(cnt, 1u);
    while (atomicAdd(cnt, 0u) < e * (u32)nwg) __builtin_amdgcn_s_sleep(1);
    __threadfence();
  }
  __syncthreads();
}

__global__ void k_lstm(u16* __restrict__ xt, u16* __restrict__ h, float* __restrict__ c,
                       float* __restrict__ gates,
                       const u16* __restrict__ Wih, const u16* __restrict__ Whh,
                       const float* __restrict__ bcomb,
                       const u16* __restrict__ WoutP, const float* __restrict__ boutP,
                       const u16* __restrict__ embbf,
                       float* __restrict__ logitsP, float* __restrict__ out,
                       u32* cnt, int nwg, int nsteps) {
  const int lane = threadIdx.x & 31;
  const int wave = threadIdx.x >> 5;
  const int g = blockIdx.x * 8 + wave;               // 0..127
  const int tid = blockIdx.x * blockDim.x + threadIdx.x;
  u32 epoch = 0;

  // Per-wave tile assignment for the gates GEMM: m-tile (g&3), 4 n-tiles (g>>2)*4+j.
  const int m0A   = (g & 3) * 16;
  const int arowA = m0A + (lane & 15);
  const int ntbA  = g >> 2;                          // 0..31
  const u16* w1p[4];
  const u16* w2p[4];
  float bvA[4];
#pragma unroll
  for (int j = 0; j < 4; ++j) {
    int ncol = (ntbA * 4 + j) * 16 + (lane & 15);
    w1p[j] = Wih + (long)ncol * 1024;
    w2p[j] = Whh + (long)ncol * 512;
    bvA[j] = bcomb[ncol];
  }

  // Logits GEMM: 76 wave-tasks, each a 16x32 tile pair (2 n-tiles share one A frag).
  const int taskC  = g;                              // valid when < 76
  const int m0C    = (taskC & 3) * 16;
  const int n0C    = (taskC >> 2) * 32;              // 0..576
  const int ncolC  = n0C + (lane & 15);

  for (int t = 1; t <= nsteps; ++t) {
    // -- Phase A: gates[64,2048] = xt[64,1024]@Wih^T + h[64,512]@Whh^T + bias
    {
      const u16* xrow = xt + arowA * 1024;
      const u16* hrow = h + arowA * 512;
      v8f z = {0, 0, 0, 0, 0, 0, 0, 0};
      v8f acc[4] = {z, z, z, z};
#pragma unroll 4
      for (int k0 = 0; k0 < 1024; k0 += 32) {
        Frag a, bb[4];
        load_a_frag(a, xrow + k0, lane);
#pragma unroll
        for (int j = 0; j < 4; ++j) load_b_frag(bb[j], w1p[j] + k0, lane);
        __builtin_prefetch((const void*)(w1p[0] + k0 + 256), 0, 1);
#pragma unroll
        for (int j = 0; j < 4; ++j) WMMA_BF16(acc[j], a, bb[j]);
      }
#pragma unroll 4
      for (int k0 = 0; k0 < 512; k0 += 32) {
        Frag a, bb[4];
        load_a_frag(a, hrow + k0, lane);
#pragma unroll
        for (int j = 0; j < 4; ++j) load_b_frag(bb[j], w2p[j] + k0, lane);
        __builtin_prefetch((const void*)(w2p[0] + k0 + 256), 0, 1);
#pragma unroll
        for (int j = 0; j < 4; ++j) WMMA_BF16(acc[j], a, bb[j]);
      }
      const int rbase = (lane >> 4) * 8;
#pragma unroll
      for (int j = 0; j < 4; ++j) {
        const int ncol = (ntbA * 4 + j) * 16 + (lane & 15);
#pragma unroll
        for (int i = 0; i < 8; ++i)
          gates[(m0A + rbase + i) * 2048 + ncol] = acc[j][i] + bvA[j];
      }
    }
    grid_sync(cnt, &epoch, nwg);

    // -- Phase B: cell update (PyTorch gate order i,f,g,o)
    for (int e = tid; e < 64 * 512; e += nwg * 256) {
      int b = e >> 9, n = e & 511;
      const float* gr = gates + b * 2048;
      float ig = gr[n], fg = gr[512 + n], gg = gr[1024 + n], og = gr[1536 + n];
      float cc = sigf(fg) * c[e] + sigf(ig) * tanhf(gg);
      c[e] = cc;
      h[e] = f2bf(sigf(og) * tanhf(cc));
    }
    grid_sync(cnt, &epoch, nwg);

    // -- Phase C: logits[64,608] = h @ WoutP^T + boutP  (cols 600..607 padded -1e30)
    if (taskC < 76) {
      const u16* hrow = h + (m0C + (lane & 15)) * 512;
      const u16* wr0 = WoutP + (long)ncolC * 512;
      const u16* wr1 = wr0 + (long)16 * 512;
      v8f acc0 = {0, 0, 0, 0, 0, 0, 0, 0};
      v8f acc1 = {0, 0, 0, 0, 0, 0, 0, 0};
#pragma unroll 4
      for (int k0 = 0; k0 < 512; k0 += 32) {
        Frag a, b0, b1;
        load_a_frag(a, hrow + k0, lane);
        load_b_frag(b0, wr0 + k0, lane);
        load_b_frag(b1, wr1 + k0, lane);
        WMMA_BF16(acc0, a, b0);
        WMMA_BF16(acc1, a, b1);
      }
      const float bv0 = boutP[ncolC];
      const float bv1 = boutP[ncolC + 16];
      const int rbase = (lane >> 4) * 8;
#pragma unroll
      for (int i = 0; i < 8; ++i) {
        int m = m0C + rbase + i;
        float v0 = acc0[i] + bv0;
        float v1 = acc1[i] + bv1;
        logitsP[m * 608 + ncolC]      = v0;
        logitsP[m * 608 + ncolC + 16] = v1;
        float* orow = out + (long)m * 390000 + (long)t * 600;
        if (ncolC < 600)      orow[ncolC]      = v0;
        if (ncolC + 16 < 600) orow[ncolC + 16] = v1;
      }
    }
    grid_sync(cnt, &epoch, nwg);

    // -- Phase D: per-row argmax (first-max) + embedding feedback into xt
    if (g < 64) {
      const int row = g;
      float best = -3.4e38f;
      int bidx = 0;
#pragma unroll
      for (int it = 0; it < 19; ++it) {            // 19*32 == 608
        int col = it * 32 + lane;
        float v = logitsP[row * 608 + col];
        if (v > best) { best = v; bidx = col; }
      }
#pragma unroll
      for (int s = 16; s > 0; s >>= 1) {
        float ov = __shfl_xor(best, s, 32);
        int   oi = __shfl_xor(bidx, s, 32);
        if (ov > best || (ov == best && oi < bidx)) { best = ov; bidx = oi; }
      }
      const u16* ep = embbf + (long)bidx * 512;
#pragma unroll 4
      for (int e = lane; e < 512; e += 32)
        xt[row * 1024 + 512 + e] = ep[e];
    }
    grid_sync(cnt, &epoch, nwg);
  }
}

// ------------------------------------------------------------------- launcher
extern "C" void kernel_launch(void* const* d_in, const int* in_sizes, int n_in,
                              void* d_out, int out_size, void* d_ws, size_t ws_size,
                              hipStream_t stream) {
  (void)in_sizes; (void)n_in; (void)out_size; (void)ws_size;
  const float* x    = (const float*)d_in[0];
  const float* cw1  = (const float*)d_in[1];
  const float* cw2  = (const float*)d_in[2];
  const float* cw3  = (const float*)d_in[3];
  const float* cw4  = (const float*)d_in[4];
  const float* cw5  = (const float*)d_in[5];
  const float* cb1  = (const float*)d_in[6];
  const float* cb2  = (const float*)d_in[7];
  const float* cb3  = (const float*)d_in[8];
  const float* cb4  = (const float*)d_in[9];
  const float* cb5  = (const float*)d_in[10];
  const float* emb  = (const float*)d_in[11];
  const float* Wih  = (const float*)d_in[12];
  const float* Whh  = (const float*)d_in[13];
  const float* bih  = (const float*)d_in[14];
  const float* bhh  = (const float*)d_in[15];
  const float* Wout = (const float*)d_in[16];
  const float* bout = (const float*)d_in[17];
  float* out = (float*)d_out;

  char* ws = (char*)d_ws;
  size_t o = 0;
  auto take = [&](size_t bytes) { size_t r = o; o += (bytes + 255) & ~(size_t)255; return r; };
  u16*   bufA   = (u16*)(ws + take(46022656ull * 2)); // conv outputs (max: L2)
  u16*   bufB   = (u16*)(ws + take(24780800ull * 2)); // pooled outputs (max: L1)
  u16*   w2t    = (u16*)(ws + take(51200ull * 2));
  u16*   w3t    = (u16*)(ws + take(204800ull * 2));
  u16*   w4t    = (u16*)(ws + take(819200ull * 2));
  u16*   w5t    = (u16*)(ws + take(3276800ull * 2));
  u16*   Wihb   = (u16*)(ws + take(2097152ull * 2));
  u16*   Whhb   = (u16*)(ws + take(1048576ull * 2));
  u16*   WoutP  = (u16*)(ws + take(311296ull * 2));
  u16*   embbf  = (u16*)(ws + take(307200ull * 2));
  float* bcomb  = (float*)(ws + take(2048ull * 4));
  float* boutP  = (float*)(ws + take(608ull * 4));
  float* ctx    = (float*)(ws + take(32768ull * 4));
  u16*   xt     = (u16*)(ws + take(65536ull * 2));
  u16*   hbuf   = (u16*)(ws + take(32768ull * 2));
  float* cbuf   = (float*)(ws + take(32768ull * 4));
  float* gates  = (float*)(ws + take(131072ull * 4));
  float* logitsP= (float*)(ws + take(38912ull * 4));
  u32*   cnt    = (u32*)(ws + take(256));

  hipMemsetAsync(cnt, 0, 4, stream);

  // weight / bias prep
  k_convw<<<(51200 + 255) / 256, 256, 0, stream>>>(cw2, w2t, 32, 51200);
  k_convw<<<(204800 + 255) / 256, 256, 0, stream>>>(cw3, w3t, 64, 204800);
  k_convw<<<(819200 + 255) / 256, 256, 0, stream>>>(cw4, w4t, 128, 819200);
  k_convw<<<(3276800 + 255) / 256, 256, 0, stream>>>(cw5, w5t, 256, 3276800);
  k_f32_to_bf16<<<(2097152 + 255) / 256, 256, 0, stream>>>(Wih, Wihb, 2097152);
  k_f32_to_bf16<<<(1048576 + 255) / 256, 256, 0, stream>>>(Whh, Whhb, 1048576);
  k_f32_to_bf16<<<(307200 + 255) / 256, 256, 0, stream>>>(emb, embbf, 307200);
  k_woutpad<<<(608 * 512 + 255) / 256, 256, 0, stream>>>(Wout, WoutP);
  k_biascomb<<<8, 256, 0, stream>>>(bih, bhh, bcomb);
  k_boutpad<<<3, 256, 0, stream>>>(bout, boutP);

  // CNN encoder
  k_conv1_pool<<<(64 * 110 * 110 + 255) / 256, 256, 0, stream>>>(x, cw1, cb1, bufB);

  k_conv_wmma<110, 110, 32, 64><<<dim3(11236, 1), 256, 0, stream>>>(bufB, w2t, cb2, bufA);
  k_pool<<<(11503616 + 255) / 256, 256, 0, stream>>>(bufA, bufB, 106, 106, 64, 53, 53, 11503616);

  k_conv_wmma<53, 53, 64, 128><<<dim3(2401, 2), 256, 0, stream>>>(bufB, w3t, cb3, bufA);
  k_pool<<<(4718592 + 255) / 256, 256, 0, stream>>>(bufA, bufB, 49, 49, 128, 24, 24, 4718592);

  k_conv_wmma<24, 24, 128, 256><<<dim3(400, 4), 256, 0, stream>>>(bufB, w4t, cb4, bufA);
  k_pool<<<(1638400 + 255) / 256, 256, 0, stream>>>(bufA, bufB, 20, 20, 256, 10, 10, 1638400);

  k_conv_wmma<10, 10, 256, 512><<<dim3(36, 8), 256, 0, stream>>>(bufB, w5t, cb5, bufA);
  k_pool<<<(294912 + 255) / 256, 256, 0, stream>>>(bufA, bufB, 6, 6, 512, 3, 3, 294912);

  k_ctx<<<128, 256, 0, stream>>>(bufB, ctx);
  k_init<<<128, 256, 0, stream>>>(ctx, embbf, xt, hbuf, cbuf);
  k_zero_step0<<<(64 * 600 + 255) / 256, 256, 0, stream>>>(out);

  // persistent LSTM decode: 649 steps
  k_lstm<<<16, 256, 0, stream>>>(xt, hbuf, cbuf, gates, Wihb, Whhb, bcomb,
                                 WoutP, boutP, embbf, logitsP, out, cnt, 16, 649);
}
